// LiquidNeuralNetwork_21586505630028
// MI455X (gfx1250) — matrix operand order repaired and checked by reading, not compiled, not run on hardware
//
#include <hip/hip_runtime.h>

// Liquid NN scan, fused single-pass kernel for MI455X (gfx1250, wave32, WMMA).
// Transposed recurrence pre^T = W^T x^T + W_rec^T h^T: WMMA C-layout (hidden
// rows x batch cols) publishes h with one ds_store_b128/lane/step; B-fragments
// load contiguously. x stream is software-pipelined one step ahead and the
// next-step cvt is hidden inside a split workgroup barrier.
// B=4096, T=512, IN=32, H=64, OUT=1.

#define BATCH  4096
#define TSTEPS 512
#define NIN    32
#define NH     64
#define HST    72   // padded bf16 LDS row stride (36 DW -> perfect bank permutation)
#define FST    68   // padded f32 LDS row stride for final h

typedef __attribute__((ext_vector_type(16))) __bf16 v16bf;
typedef __attribute__((ext_vector_type(8)))  __bf16 v8bf;
typedef __attribute__((ext_vector_type(8)))  float  v8f;
typedef __attribute__((ext_vector_type(4)))  float  v4f;

#if __has_builtin(__builtin_amdgcn_tanhf)
#define FAST_TANH(v) __builtin_amdgcn_tanhf(v)
#else
#define FAST_TANH(v) tanhf(v)
#endif

#if __has_builtin(__builtin_amdgcn_s_barrier_signal) && __has_builtin(__builtin_amdgcn_s_barrier_wait)
#define HAVE_SPLIT_BARRIER 1
#endif

__global__ __launch_bounds__(128)
void lnn_scan_kernel(const float* __restrict__ x,
                     const float* __restrict__ W_in,
                     const float* __restrict__ b_in,
                     const float* __restrict__ W_rec,
                     const float* __restrict__ b_rec,
                     const float* __restrict__ tau,
                     const float* __restrict__ W_fc,
                     const float* __restrict__ b_fc,
                     float* __restrict__ out)
{
    // ping-pong bf16 h^T [batch 16][hidden 64] for WMMA B-operand
    __shared__ __align__(16) __bf16 hbuf[2][16 * HST];
    __shared__ float hfin[16 * FST];

    const int tid   = threadIdx.x;
    const int lane  = tid & 31;
    const int wave  = tid >> 5;        // 0..3 -> hidden-row tile of 16
    const int n0    = wave * 16;       // hidden-row tile base
    const int b0    = blockIdx.x * 16; // 16 batch rows per block

    const int m      = lane & 15;      // A row (hidden within tile) AND B/C col (batch row)
    const int hiH    = lane >> 4;      // lane half
    const int kbA    = hiH * 8;        // unused (table order would be i<8 -> kbA+i, i>=8 -> kbA+8+i; A uses klB+i)
    const int klB    = hiH * 16;       // B-frag K base (elem i -> klB+i, contiguous)
    const int rowoff = hiH * 8;        // C/D row offset (VGPR r -> M = r+rowoff)

    // ---- loop-invariant A fragments: transposed weights, cast f32 -> bf16 once ----
    v16bf aWin, aWr0, aWr1;
#pragma unroll
    for (int i = 0; i < 16; ++i) {
        const int k = klB + i;     // A-frag K mapping (16-bit A 16x32)
        aWin[i] = (__bf16)W_in [k        * NH + (n0 + m)];  // W_in^T [hidden][input]
        aWr0[i] = (__bf16)W_rec[k        * NH + (n0 + m)];  // W_rec^T, K = 0..31
        aWr1[i] = (__bf16)W_rec[(k + 32) * NH + (n0 + m)];  // W_rec^T, K = 32..63
    }

    // per-element bias / 1/tau vectors (hidden index = n0 + rowoff + r)
    v8f biasv, rtauv;
#pragma unroll
    for (int r = 0; r < 8; ++r) {
        const int hidx = n0 + rowoff + r;
        biasv[r] = b_in[hidx] + b_rec[hidx];
        rtauv[r] = 1.0f / tau[hidx];
    }

    // zero initial h (only buffer 0 is read at t=0; buffer 1 is fully written at t=0)
    for (int i = tid; i < 16 * HST; i += 128)
        hbuf[0][i] = (__bf16)0.0f;
    __syncthreads();

    v8f h = {};   // fp32 recurrent state: 8 hidden values of batch row (b0+m)

    // x^T B-fragment source: x[b0+m, t, klB .. klB+15] (16 contiguous floats)
    const float* xrow = x + (size_t)(b0 + m) * TSTEPS * NIN + klB;

    // ---- software pipeline: preload + convert x fragment for t=0 ----
    v4f x0 = *(const v4f*)(xrow + 0);
    v4f x1 = *(const v4f*)(xrow + 4);
    v4f x2 = *(const v4f*)(xrow + 8);
    v4f x3 = *(const v4f*)(xrow + 12);
    v16bf bX;
#pragma unroll
    for (int i = 0; i < 4; ++i) {
        bX[i] = (__bf16)x0[i]; bX[4+i] = (__bf16)x1[i];
        bX[8+i] = (__bf16)x2[i]; bX[12+i] = (__bf16)x3[i];
    }

    int rb = 0;
    for (int t = 0; t < TSTEPS; ++t) {
        const __bf16* rbuf = hbuf[rb];
        __bf16*       wbuf = hbuf[rb ^ 1];

        // ---- B fragments of h^T (K=64 -> two 32x16 frags), 4x ds_load_b128 ----
        const int hb = m * HST;
        v8bf h0a = *(const v8bf*)&rbuf[hb + klB];
        v8bf h0b = *(const v8bf*)&rbuf[hb + klB + 8];
        v8bf h1a = *(const v8bf*)&rbuf[hb + 32 + klB];
        v8bf h1b = *(const v8bf*)&rbuf[hb + 32 + klB + 8];
        v16bf bH0 = __builtin_shufflevector(h0a, h0b, 0,1,2,3,4,5,6,7,8,9,10,11,12,13,14,15);
        v16bf bH1 = __builtin_shufflevector(h1a, h1b, 0,1,2,3,4,5,6,7,8,9,10,11,12,13,14,15);

        // ---- issue x loads for t+1 now (in flight across WMMAs + barrier) ----
        const int tn = (t + 1 < TSTEPS) ? (t + 1) : (TSTEPS - 1);  // clamp: no OOB
        const float* xn = xrow + (size_t)tn * NIN;
        v4f nx0 = *(const v4f*)(xn + 0);
        v4f nx1 = *(const v4f*)(xn + 4);
        v4f nx2 = *(const v4f*)(xn + 8);
        v4f nx3 = *(const v4f*)(xn + 12);
        __builtin_prefetch(xn + 8 * NIN, 0, 1);   // and stream 8 steps ahead

        // ---- pre^T = bias + W_in^T x^T + W_rec^T h^T, fp32 accumulate ----
        v8f acc = biasv;
        acc = __builtin_amdgcn_wmma_f32_16x16x32_bf16(false, aWin, false, bX,  (short)0, acc, false, false);
        acc = __builtin_amdgcn_wmma_f32_16x16x32_bf16(false, aWr0, false, bH0, (short)0, acc, false, false);
        acc = __builtin_amdgcn_wmma_f32_16x16x32_bf16(false, aWr1, false, bH1, (short)0, acc, false, false);

        // ---- liquid update in fp32; publish 8 contiguous bf16 with ONE ds_store_b128 ----
        v8bf hpub;
#pragma unroll
        for (int r = 0; r < 8; ++r) {
            float th = FAST_TANH(acc[r]);
            float hn = h[r] + (th - h[r]) * rtauv[r];
            h[r]    = hn;
            hpub[r] = (__bf16)hn;
        }
        *(v8bf*)&wbuf[m * HST + n0 + rowoff] = hpub;

#ifdef HAVE_SPLIT_BARRIER
        // split barrier: signal after LDS publish, convert next x while waiting
        asm volatile("" ::: "memory");
        asm volatile("s_wait_dscnt 0x0" ::: "memory");  // h publish visible pre-signal
        __builtin_amdgcn_s_barrier_signal(-1);
#pragma unroll
        for (int i = 0; i < 4; ++i) {
            bX[i] = (__bf16)nx0[i]; bX[4+i] = (__bf16)nx1[i];
            bX[8+i] = (__bf16)nx2[i]; bX[12+i] = (__bf16)nx3[i];
        }
        __builtin_amdgcn_s_barrier_wait(-1);
        asm volatile("" ::: "memory");
#else
#pragma unroll
        for (int i = 0; i < 4; ++i) {
            bX[i] = (__bf16)nx0[i]; bX[4+i] = (__bf16)nx1[i];
            bX[8+i] = (__bf16)nx2[i]; bX[12+i] = (__bf16)nx3[i];
        }
        __syncthreads();
#endif
        rb ^= 1;
    }

    // ---- epilogue: out[b] = h[b,:] @ W_fc + b_fc (deterministic, no atomics) ----
    {
        v4f hlo = { h[0], h[1], h[2], h[3] };
        v4f hhi = { h[4], h[5], h[6], h[7] };
        float* fp = &hfin[m * FST + n0 + rowoff];
        *(v4f*)(fp)     = hlo;
        *(v4f*)(fp + 4) = hhi;
    }
    __syncthreads();

    if (tid < 16) {
        float s = b_fc[0];
        const float* hr = &hfin[tid * FST];
#pragma unroll 8
        for (int c = 0; c < NH; ++c)
            s += hr[c] * W_fc[c];
        out[b0 + tid] = s;
    }
}

extern "C" void kernel_launch(void* const* d_in, const int* in_sizes, int n_in,
                              void* d_out, int out_size, void* d_ws, size_t ws_size,
                              hipStream_t stream) {
    const float* x     = (const float*)d_in[0];
    const float* W_in  = (const float*)d_in[1];
    const float* b_in  = (const float*)d_in[2];
    const float* W_rec = (const float*)d_in[3];
    const float* b_rec = (const float*)d_in[4];
    const float* tau   = (const float*)d_in[5];
    const float* W_fc  = (const float*)d_in[6];
    const float* b_fc  = (const float*)d_in[7];
    float* out = (float*)d_out;

    dim3 grid(BATCH / 16);   // 256 blocks x 4 waves = 1024 WMMA waves
    dim3 block(128);
    hipLaunchKernelGGL(lnn_scan_kernel, grid, block, 0, stream,
                       x, W_in, b_in, W_rec, b_rec, tau, W_fc, b_fc, out);
}
